// HANModel_88974542504580
// MI455X (gfx1250) — matrix operand order repaired and checked by reading, not compiled, not run on hardware
//
#include <hip/hip_runtime.h>
#include <hip/hip_bf16.h>

// Problem constants (from reference)
#define NE 100000
#define NU 200000
#define NS 20000
#define EU 500000
#define ES 500000
#define KSTEPS 24   // 768 / 32

typedef __attribute__((ext_vector_type(16))) __bf16 v16bf;
typedef __attribute__((ext_vector_type(8)))  float  v8f;
typedef __attribute__((ext_vector_type(4)))  unsigned int v4u;
typedef __attribute__((ext_vector_type(8)))  int  v8i;
typedef __attribute__((ext_vector_type(4)))  int  v4i;

__device__ __forceinline__ float waveSum(float v) {
#pragma unroll
  for (int off = 16; off > 0; off >>= 1) v += __shfl_xor(v, off, 32);
  return v;
}

// float atomic max via int punning (works for mixed signs, init = -inf)
__device__ __forceinline__ void atomicMaxF(float* addr, float val) {
  if (val >= 0.0f) atomicMax((int*)addr, __float_as_int(val));
  else             atomicMin((unsigned int*)addr, __float_as_uint(val));
}

// ------------------------------------------------------------------ fill
__global__ void k_fill(float* __restrict__ p, float v, long n) {
  long i = (long)blockIdx.x * blockDim.x + threadIdx.x;
  if (i < n) p[i] = v;
}

__global__ void k_relu(float* __restrict__ p, long n) {
  long i = (long)blockIdx.x * blockDim.x + threadIdx.x;
  if (i < n) p[i] = fmaxf(p[i], 0.0f);
}

// --------------------------------------------- affine fusion: Wf=Wp@Wl, bf=bp@Wl+bl
__global__ void k_fuse_affine(const float* __restrict__ Wp, const float* __restrict__ bp,
                              const float* __restrict__ Wl, const float* __restrict__ bl,
                              float* __restrict__ Wf, float* __restrict__ bf, int Kin) {
  int idx = blockIdx.x * blockDim.x + threadIdx.x;
  if (idx >= Kin * 128) return;
  int i = idx >> 7, j = idx & 127;
  float s = 0.f;
  for (int k = 0; k < 128; ++k) s += Wp[i * 128 + k] * Wl[k * 128 + j];
  Wf[i * 128 + j] = s;
  if (i == 0) {
    float b = bl[j];
    for (int k = 0; k < 128; ++k) b += bp[k] * Wl[k * 128 + j];
    bf[j] = b;
  }
}

// -------------------------------- pack B [768,128] f32 -> bf16 WMMA fragment order
// layout: [kstep(24)][ntile(8)][lane(32)][elem(16)]
// B 32x16 bf16 fragment, contiguous k order used here: lanes 0-15 hold K=0..15 (elem i -> K=i), lanes 16-31 K=16..31
__global__ void k_pack_B(const float* __restrict__ Wf, __bf16* __restrict__ Bp) {
  int idx = blockIdx.x * blockDim.x + threadIdx.x;  // 24*8*32 = 6144
  if (idx >= KSTEPS * 8 * 32) return;
  int lane = idx & 31;
  int nt   = (idx >> 5) & 7;
  int ks   = idx >> 8;
  int n  = nt * 16 + (lane & 15);
  int kb = ks * 32 + ((lane >> 4) << 4);
  __bf16* o = Bp + (long)idx * 16;
#pragma unroll
  for (int i = 0; i < 16; ++i) o[i] = (__bf16)Wf[(long)(kb + i) * 128 + n];
}

// ------------------------------------- big GEMM: z_e = x_email @ Wf_e + bf_e (WMMA)
// Block = 8 waves x 16 rows = 128 rows. Packed B (192KB bf16) is DMA'd once per
// block into LDS by the Tensor Data Mover, then all waves read fragments via ds_load.
// A is streamed f32 from HBM (read exactly once) and converted to bf16 in registers.
// B fragments are rotation-pipelined: fragment t+1 is loaded from LDS before the
// WMMA consuming fragment t issues, forcing true double-buffering in regalloc so
// each LDS round-trip hides under one WMMA.
__global__ __launch_bounds__(256)
void k_gemm_email(const float* __restrict__ X,    // [NE,768]
                  const __bf16* __restrict__ Bp,  // packed fragments (global)
                  const float* __restrict__ bias, // [128] fused bias
                  float* __restrict__ Z) {        // [NE,128]
  extern __shared__ __bf16 Bl[];                  // 768*128 bf16 = 196608 B
  const int lane = threadIdx.x & 31;
  const int wave = threadIdx.x >> 5;
  const int rowBase = (blockIdx.x * 8 + wave) * 16;

  // ---- TDM: wave 0 issues a 1-D tensor DMA of the whole packed B into LDS ----
  if (wave == 0) {
    unsigned long long ga = (unsigned long long)Bp;
    // D# group 0: count=1 | lds_addr=0 (dynamic-LDS base) | global_addr | type=2
    v4u g0;
    g0[0] = 1u;
    g0[1] = 0u;
    g0[2] = (unsigned)(ga & 0xFFFFFFFFu);
    g0[3] = (unsigned)((ga >> 32) & 0x1FFFFFFu) | (2u << 30);
    // D# group 1: data_size=8B; tensor_dim0 = tile_dim0 = stride = 24576 (x8B = 192KB)
    v8i g1;
    g1[0] = 0x00030000;              // data_size=3 (8 bytes)
    g1[1] = (int)(0x6000u << 16);    // tensor_dim0[15:0] = 24576 at bits 63:48
    g1[2] = 0x00010000;              // tensor_dim1 = 1 at bits 95:80
    g1[3] = (int)(0x6000u << 16);    // tile_dim0 = 24576 at bits 127:112
    g1[4] = 1;                       // tile_dim1 = 1
    g1[5] = 24576;                   // tensor_dim0_stride
    g1[6] = 0; g1[7] = 0;
    v4i g2 = {0, 0, 0, 0};
    v4i g3 = {0, 0, 0, 0};
    v8i g4 = {0, 0, 0, 0, 0, 0, 0, 0};
    __builtin_amdgcn_tensor_load_to_lds(g0, g1, g2, g3, g4, 0);
    __builtin_amdgcn_s_wait_tensorcnt(0);
  }
  __syncthreads();
  if (rowBase >= NE) return;                 // after barrier: safe wave-uniform exit

  const int half = lane >> 4;
  const int l16  = lane & 15;
  const int m    = rowBase + l16;
  // A 16x32 bf16 k order used here (contiguous, matches packed B): lanes 0-15: K {0..15}; lanes 16-31: K {16..31}
  const int kb0 = half ? 16 : 0;
  const int kb1 = half ? 24 : 8;

  v8f acc[8] = {};
  const float* xrow = X + (long)m * 768;

  for (int ks = 0; ks < KSTEPS; ++ks) {
    // A fragment: two contiguous 32B f32 chunks -> bf16 in registers
    const float* p0 = xrow + ks * 32 + kb0;
    const float* p1 = xrow + ks * 32 + kb1;
    float4 A0 = ((const float4*)p0)[0], A1 = ((const float4*)p0)[1];
    float4 B0 = ((const float4*)p1)[0], B1 = ((const float4*)p1)[1];
    v16bf af;
    af[0]=(__bf16)A0.x; af[1]=(__bf16)A0.y; af[2]=(__bf16)A0.z; af[3]=(__bf16)A0.w;
    af[4]=(__bf16)A1.x; af[5]=(__bf16)A1.y; af[6]=(__bf16)A1.z; af[7]=(__bf16)A1.w;
    af[8]=(__bf16)B0.x; af[9]=(__bf16)B0.y; af[10]=(__bf16)B0.z; af[11]=(__bf16)B0.w;
    af[12]=(__bf16)B1.x; af[13]=(__bf16)B1.y; af[14]=(__bf16)B1.z; af[15]=(__bf16)B1.w;

    // rotation pipeline over the 8 B fragments of this k-step
    const __bf16* bbase = Bl + ((long)ks * 8) * 32 * 16 + lane * 16;
    v16bf bcur = *(const v16bf*)(bbase);
#pragma unroll
    for (int t = 0; t < 7; ++t) {
      v16bf bnxt = *(const v16bf*)(bbase + (long)(t + 1) * 32 * 16);
      acc[t] = __builtin_amdgcn_wmma_f32_16x16x32_bf16(
          false, af, false, bcur, (short)0, acc[t], false, false);
      bcur = bnxt;
    }
    acc[7] = __builtin_amdgcn_wmma_f32_16x16x32_bf16(
        false, af, false, bcur, (short)0, acc[7], false, false);
  }
  // epilogue: +bias, store. C/D: VGPR r -> row rowBase + half*8 + r, col = t*16+l16
#pragma unroll
  for (int t = 0; t < 8; ++t) {
    int n = t * 16 + l16;
    float b = bias[n];
#pragma unroll
    for (int r = 0; r < 8; ++r) {
      int row = rowBase + half * 8 + r;
      Z[(long)row * 128 + n] = acc[t][r] + b;
    }
  }
}

// ----------------------------- skinny GEMMs (K=8 / K=1): thread per output element
__global__ void k_gemm_small(const float* __restrict__ X, const float* __restrict__ Wf,
                             const float* __restrict__ bf, float* __restrict__ Z,
                             int N, int Kin) {
  long idx = (long)blockIdx.x * blockDim.x + threadIdx.x;
  if (idx >= (long)N * 128) return;
  int n = (int)(idx >> 7), j = (int)(idx & 127);
  float s = bf[j];
  for (int k = 0; k < Kin; ++k) s += X[(long)n * Kin + k] * Wf[k * 128 + j];
  Z[idx] = s;
}

// ----------------------- per-node attention logits: s[n,h] = sum_d z[n,h,d]*att[h,d]
__global__ __launch_bounds__(256)
void k_node_scores(const float* __restrict__ z, const float* __restrict__ att,
                   float* __restrict__ s, int N) {
  int n = blockIdx.x * 8 + (threadIdx.x >> 5);
  if (n >= N) return;
  int lane = threadIdx.x & 31;
  const float* zr = z + (long)n * 128;
  float p0 = zr[lane] * att[lane] + zr[lane + 32] * att[lane + 32];
  float p1 = zr[lane + 64] * att[lane + 64] + zr[lane + 96] * att[lane + 96];
  p0 = waveSum(p0); p1 = waveSum(p1);
  if (lane == 0) { s[n * 2] = p0; s[n * 2 + 1] = p1; }
}

// ------------------------------------------- edge pass 1: alpha + segment max
__global__ void k_edge_alpha(const float* __restrict__ ssrc, const float* __restrict__ sdst,
                             const int* __restrict__ src, const int* __restrict__ dst,
                             float* __restrict__ abuf, float* __restrict__ m, int E) {
  int e = blockIdx.x * blockDim.x + threadIdx.x;
  if (e >= E) return;
  int si = src[e], di = dst[e];
#pragma unroll
  for (int h = 0; h < 2; ++h) {
    float a = ssrc[si * 2 + h] + sdst[di * 2 + h];
    a = a > 0.f ? a : 0.2f * a;   // leaky_relu(0.2)
    abuf[(long)e * 2 + h] = a;
    atomicMaxF(&m[di * 2 + h], a);
  }
}

// ------------------------------------------- edge pass 2: exp + segment sum
__global__ void k_edge_exp(float* __restrict__ abuf, const float* __restrict__ m,
                           const int* __restrict__ dst, float* __restrict__ ssum, int E) {
  int e = blockIdx.x * blockDim.x + threadIdx.x;
  if (e >= E) return;
  int di = dst[e];
#pragma unroll
  for (int h = 0; h < 2; ++h) {
    float ex = expf(abuf[(long)e * 2 + h] - m[di * 2 + h]);
    abuf[(long)e * 2 + h] = ex;
    atomicAdd(&ssum[di * 2 + h], ex);
  }
}

// -------------------- edge pass 3: weighted scatter out[dst] += z_src[src]*a (L2 atomics)
__global__ __launch_bounds__(256)
void k_edge_scatter(const float* __restrict__ zsrc, const float* __restrict__ ebuf,
                    const float* __restrict__ ssum, const int* __restrict__ src,
                    const int* __restrict__ dst, float* __restrict__ out, int E) {
  int e = blockIdx.x * 8 + (threadIdx.x >> 5);
  if (e >= E) return;
  int lane = threadIdx.x & 31;
  int si = src[e], di = dst[e];
  float a0 = ebuf[(long)e * 2]     / (ssum[di * 2]     + 1e-16f);
  float a1 = ebuf[(long)e * 2 + 1] / (ssum[di * 2 + 1] + 1e-16f);
  const float* zr = zsrc + (long)si * 128;
  float* o = out + (long)di * 128;
#pragma unroll
  for (int i = 0; i < 4; ++i) {
    int c = lane + 32 * i;
    float a = (i < 2) ? a0 : a1;   // c<64 -> head 0
    atomicAdd(&o[c], zr[c] * a);
  }
}

// ---------------- semantic score: score[r] = sum_n q . tanh(out_r[n] @ Wk + bk)
// block handles 32 k-columns (Wk tile in LDS) x a node stripe; grid = 4 kgroups x 128
__global__ __launch_bounds__(128)
void k_semantic(const float* __restrict__ outR, const float* __restrict__ Wk,
                const float* __restrict__ bk, const float* __restrict__ q,
                float* __restrict__ scorePtr) {
  __shared__ float wkS[128 * 32];
  __shared__ float row[128];
  __shared__ float part[128];
  __shared__ float red[32];
  int t = threadIdx.x;
  int kg = blockIdx.x >> 7;        // 0..3
  int split = blockIdx.x & 127;    // 0..127
  int k0 = kg * 32;
  for (int i = t; i < 128 * 32; i += 128) {
    int c = i >> 5, kk = i & 31;
    wkS[i] = Wk[c * 128 + k0 + kk];
  }
  __syncthreads();
  int kk = t & 31, qtr = t >> 5;
  float qk = q[k0 + kk], bkk = bk[k0 + kk];
  float acc = 0.f;
  for (int n = split; n < NE; n += 128) {
    row[t] = outR[(long)n * 128 + t];
    __syncthreads();
    float p = 0.f;
#pragma unroll 8
    for (int j = 0; j < 32; ++j) p += row[qtr * 32 + j] * wkS[(qtr * 32 + j) * 32 + kk];
    part[t] = p;
    __syncthreads();
    if (t < 32) {
      float d = part[t] + part[t + 32] + part[t + 64] + part[t + 96] + bkk;
      acc += qk * tanhf(d);
    }
    __syncthreads();
  }
  if (t < 32) red[t] = acc;
  __syncthreads();
  if (t == 0) {
    float s = 0.f;
    for (int i = 0; i < 32; ++i) s += red[i];
    atomicAdd(scorePtr, s);
  }
}

// --------------------------------------------------------- beta = softmax(score/NE)
__global__ void k_beta(const float* __restrict__ score, float* __restrict__ beta) {
  float s0 = score[0] / (float)NE, s1 = score[1] / (float)NE;
  float mx = fmaxf(s0, s1);
  float e0 = expf(s0 - mx), e1 = expf(s1 - mx);
  float inv = 1.f / (e0 + e1);
  beta[0] = e0 * inv; beta[1] = e1 * inv;
}

// ------------------------------- final: out = (b0*out_ue + b1*out_se) @ Wc + bc
__global__ __launch_bounds__(256)
void k_final(const float* __restrict__ oUE, const float* __restrict__ oSE,
             const float* __restrict__ beta, const float* __restrict__ Wc,
             const float* __restrict__ bc, float* __restrict__ out) {
  int n = blockIdx.x * 8 + (threadIdx.x >> 5);
  if (n >= NE) return;
  int lane = threadIdx.x & 31;
  float b0 = beta[0], b1 = beta[1];
  float p0 = 0.f, p1 = 0.f;
#pragma unroll
  for (int i = 0; i < 4; ++i) {
    int c = lane + 32 * i;
    float f = b0 * oUE[(long)n * 128 + c] + b1 * oSE[(long)n * 128 + c];
    p0 += f * Wc[c * 2];
    p1 += f * Wc[c * 2 + 1];
  }
  p0 = waveSum(p0); p1 = waveSum(p1);
  if (lane == 0) { out[n * 2] = p0 + bc[0]; out[n * 2 + 1] = p1 + bc[1]; }
}

// =====================================================================
extern "C" void kernel_launch(void* const* d_in, const int* in_sizes, int n_in,
                              void* d_out, int out_size, void* d_ws, size_t ws_size,
                              hipStream_t stream) {
  const float* x_email  = (const float*)d_in[0];
  const float* x_url    = (const float*)d_in[1];
  const float* x_sender = (const float*)d_in[2];
  const float* Wp_e = (const float*)d_in[3];  const float* bp_e = (const float*)d_in[4];
  const float* Wp_u = (const float*)d_in[5];  const float* bp_u = (const float*)d_in[6];
  const float* Wp_s = (const float*)d_in[7];  const float* bp_s = (const float*)d_in[8];
  const float* Wl_e = (const float*)d_in[9];  const float* bl_e = (const float*)d_in[10];
  const float* Wl_u = (const float*)d_in[11]; const float* bl_u = (const float*)d_in[12];
  const float* Wl_s = (const float*)d_in[13]; const float* bl_s = (const float*)d_in[14];
  const float* att_src_ue = (const float*)d_in[15];
  const float* att_dst_ue = (const float*)d_in[16];
  const float* att_src_se = (const float*)d_in[17];
  const float* att_dst_se = (const float*)d_in[18];
  const float* Wk = (const float*)d_in[19]; const float* bk = (const float*)d_in[20];
  const float* q  = (const float*)d_in[21];
  const float* Wc = (const float*)d_in[22]; const float* bc = (const float*)d_in[23];
  const int* src_ue = (const int*)d_in[24]; const int* dst_ue = (const int*)d_in[25];
  const int* src_se = (const int*)d_in[26]; const int* dst_se = (const int*)d_in[27];
  float* out = (float*)d_out;

  // workspace carve (256B aligned)
  char* w = (char*)d_ws;
  auto alloc = [&](size_t bytes) -> char* {
    char* p = w; w += (bytes + 255) & ~(size_t)255; return p;
  };
  float* Wf_e = (float*)alloc(768 * 128 * 4);
  float* bf_e = (float*)alloc(128 * 4);
  float* Wf_u = (float*)alloc(8 * 128 * 4);
  float* bf_u = (float*)alloc(128 * 4);
  float* Wf_s = (float*)alloc(128 * 4);
  float* bf_s = (float*)alloc(128 * 4);
  __bf16* Bp  = (__bf16*)alloc(768 * 128 * 2);
  float* z_e  = (float*)alloc((size_t)NE * 128 * 4);
  float* z_u  = (float*)alloc((size_t)NU * 128 * 4);
  float* z_s  = (float*)alloc((size_t)NS * 128 * 4);
  float* s_src_ue = (float*)alloc((size_t)NU * 2 * 4);
  float* s_dst_ue = (float*)alloc((size_t)NE * 2 * 4);
  float* s_src_se = (float*)alloc((size_t)NS * 2 * 4);
  float* s_dst_se = (float*)alloc((size_t)NE * 2 * 4);
  float* m_ue    = (float*)alloc((size_t)NE * 2 * 4);
  float* ssum_ue = (float*)alloc((size_t)NE * 2 * 4);
  float* m_se    = (float*)alloc((size_t)NE * 2 * 4);
  float* ssum_se = (float*)alloc((size_t)NE * 2 * 4);
  float* e_ue = (float*)alloc((size_t)EU * 2 * 4);
  float* e_se = (float*)alloc((size_t)ES * 2 * 4);
  float* out_ue = (float*)alloc((size_t)NE * 128 * 4);
  float* out_se = (float*)alloc((size_t)NE * 128 * 4);
  float* score  = (float*)alloc(2 * 4);
  float* beta   = (float*)alloc(2 * 4);

  // 1. affine fusion: z = x @ (Wp@Wl) + (bp@Wl + bl)
  k_fuse_affine<<<(768 * 128 + 255) / 256, 256, 0, stream>>>(Wp_e, bp_e, Wl_e, bl_e, Wf_e, bf_e, 768);
  k_fuse_affine<<<(8 * 128 + 255) / 256, 256, 0, stream>>>(Wp_u, bp_u, Wl_u, bl_u, Wf_u, bf_u, 8);
  k_fuse_affine<<<1, 256, 0, stream>>>(Wp_s, bp_s, Wl_s, bl_s, Wf_s, bf_s, 1);

  // 2. pack B into bf16 WMMA fragment order (192KB, TDM-staged into LDS per block)
  k_pack_B<<<(KSTEPS * 8 * 32 + 255) / 256, 256, 0, stream>>>(Wf_e, Bp);

  // 3. GEMMs (128 rows per block, 8 waves; 196608B dynamic LDS for packed B)
  k_gemm_email<<<(NE / 16 + 7) / 8, 256, 196608, stream>>>(x_email, Bp, bf_e, z_e);
  k_gemm_small<<<(int)(((long)NU * 128 + 255) / 256), 256, 0, stream>>>(x_url, Wf_u, bf_u, z_u, NU, 8);
  k_gemm_small<<<(int)(((long)NS * 128 + 255) / 256), 256, 0, stream>>>(x_sender, Wf_s, bf_s, z_s, NS, 1);

  // 4. per-node attention logits
  k_node_scores<<<(NU + 7) / 8, 256, 0, stream>>>(z_u, att_src_ue, s_src_ue, NU);
  k_node_scores<<<(NE + 7) / 8, 256, 0, stream>>>(z_e, att_dst_ue, s_dst_ue, NE);
  k_node_scores<<<(NS + 7) / 8, 256, 0, stream>>>(z_s, att_src_se, s_src_se, NS);
  k_node_scores<<<(NE + 7) / 8, 256, 0, stream>>>(z_e, att_dst_se, s_dst_se, NE);

  // 5. init accumulators
  const float NEG_INF = -__builtin_huge_valf();
  long n2 = (long)NE * 2, n128 = (long)NE * 128;
  k_fill<<<(int)((n2 + 255) / 256), 256, 0, stream>>>(m_ue, NEG_INF, n2);
  k_fill<<<(int)((n2 + 255) / 256), 256, 0, stream>>>(m_se, NEG_INF, n2);
  k_fill<<<(int)((n2 + 255) / 256), 256, 0, stream>>>(ssum_ue, 0.f, n2);
  k_fill<<<(int)((n2 + 255) / 256), 256, 0, stream>>>(ssum_se, 0.f, n2);
  k_fill<<<(int)((n128 + 255) / 256), 256, 0, stream>>>(out_ue, 0.f, n128);
  k_fill<<<(int)((n128 + 255) / 256), 256, 0, stream>>>(out_se, 0.f, n128);
  k_fill<<<1, 256, 0, stream>>>(score, 0.f, 2);

  // 6. url->email relation
  k_edge_alpha<<<(EU + 255) / 256, 256, 0, stream>>>(s_src_ue, s_dst_ue, src_ue, dst_ue, e_ue, m_ue, EU);
  k_edge_exp<<<(EU + 255) / 256, 256, 0, stream>>>(e_ue, m_ue, dst_ue, ssum_ue, EU);
  k_edge_scatter<<<(EU + 7) / 8, 256, 0, stream>>>(z_u, e_ue, ssum_ue, src_ue, dst_ue, out_ue, EU);

  // 7. sender->email relation
  k_edge_alpha<<<(ES + 255) / 256, 256, 0, stream>>>(s_src_se, s_dst_se, src_se, dst_se, e_se, m_se, ES);
  k_edge_exp<<<(ES + 255) / 256, 256, 0, stream>>>(e_se, m_se, dst_se, ssum_se, ES);
  k_edge_scatter<<<(ES + 7) / 8, 256, 0, stream>>>(z_s, e_se, ssum_se, src_se, dst_se, out_se, ES);

  // 8. relu
  k_relu<<<(int)((n128 + 255) / 256), 256, 0, stream>>>(out_ue, n128);
  k_relu<<<(int)((n128 + 255) / 256), 256, 0, stream>>>(out_se, n128);

  // 9. semantic attention
  k_semantic<<<512, 128, 0, stream>>>(out_ue, Wk, bk, q, &score[0]);
  k_semantic<<<512, 128, 0, stream>>>(out_se, Wk, bk, q, &score[1]);
  k_beta<<<1, 1, 0, stream>>>(score, beta);

  // 10. fuse + classifier
  k_final<<<(NE + 7) / 8, 256, 0, stream>>>(out_ue, out_se, beta, Wc, bc, out);
}
